// LSTM_60275571032767
// MI455X (gfx1250) — matrix operand order repaired and checked by reading, not compiled, not run on hardware
//
#include <hip/hip_runtime.h>

// LSTM scan: T=1024, B=64, D=512, gates 4D=2048.
// Persistent kernel, 16 WGs; each WG keeps its 256KB f16 slice of [Wi;Wh]^T
// in LDS for the whole scan. 128 waves each own one 16x16 output tile
// (4 gate accumulators), cell state lives in registers, device-wide barrier
// between time steps. Matmul = v_wmma_f32_16x16x32_f16, fp32 accum.
// This revision: amdgpu_waves_per_eu(2) occupancy hint (exactly our
// residency: 8 wave32 on 4 SIMD32s) so the register allocator gets the full
// per-wave VGPR budget and can keep all four next-iteration B fragments live
// (previous builds collapsed the B double-buffer into one register set,
// forcing s_wait_dscnt 0 before WMMAs 2..4 of each group).

#define T_STEPS  1024
#define BATCH    64
#define DMODEL   512
#define G4D      2048
#define NWG      16
#define NTHREADS 256

typedef __attribute__((ext_vector_type(16))) _Float16 v16h;
typedef __attribute__((ext_vector_type(8)))  _Float16 v8h;
typedef __attribute__((ext_vector_type(4)))  float    v4f;
typedef __attribute__((ext_vector_type(8)))  float    v8f;
typedef __attribute__((ext_vector_type(4)))  _Float16 v4h;

__device__ __forceinline__ float sigf(float x) {
    return 1.0f / (1.0f + __expf(-x));
}
__device__ __forceinline__ float tanh_fast(float x) {
    return 2.0f / (1.0f + __expf(-2.0f * x)) - 1.0f;
}

// A fragment (16x32 f16): lanes 0-15 take K{0..15}, lanes 16-31 take
// K{16..31} (aoff pre-applied by caller).
__device__ __forceinline__ v16h load_a_x(const float* p) {   // fallback path
    v16h a;
#pragma unroll
    for (int j = 0; j < 8; ++j) {
        a[j]     = (_Float16)p[j];
        a[8 + j] = (_Float16)p[8 + j];
    }
    return a;
}
__device__ __forceinline__ v16h load_a_h(const _Float16* p) {
    v8h lo = *(const v8h*)p;
    v8h hi = *(const v8h*)(p + 8);
    v16h a;
#pragma unroll
    for (int j = 0; j < 8; ++j) {
        a[j]     = lo[j];
        a[8 + j] = hi[j];
    }
    return a;
}

// One-time per launch: reset barrier counter, convert h0 -> f16 ping buffer.
__global__ void lstm_init_kernel(const float* __restrict__ h0,
                                 _Float16* __restrict__ hbuf0,
                                 unsigned* __restrict__ counter) {
    int idx = blockIdx.x * blockDim.x + threadIdx.x;
    if (idx == 0) *counter = 0u;
    if (idx < BATCH * DMODEL) hbuf0[idx] = (_Float16)h0[idx];
}

// One-time per launch: stream-convert the whole input tensor to f16.
__global__ void lstm_xcvt_kernel(const float* __restrict__ x,
                                 _Float16* __restrict__ x16, long n4) {
    long i = (long)blockIdx.x * blockDim.x + threadIdx.x;
    long stride = (long)gridDim.x * blockDim.x;
    for (; i < n4; i += stride) {
        v4f v = *(const v4f*)(x + i * 4);
        v4h o;
#pragma unroll
        for (int j = 0; j < 4; ++j) o[j] = (_Float16)v[j];
        *(v4h*)(x16 + i * 4) = o;
    }
}

template <bool PRECVT>
__global__ __launch_bounds__(NTHREADS, 1)
__attribute__((amdgpu_waves_per_eu(2)))
void lstm_persistent_kernel(const float* __restrict__ x,    // [T][B][D] f32
                            const _Float16* __restrict__ x16, // [T][B][D] f16
                            const float* __restrict__ c0,   // [B][D]    f32
                            const float* __restrict__ Wi,   // [D][4D]   f32
                            const float* __restrict__ Wh,   // [D][4D]   f32
                            const float* __restrict__ bias, // [4D]      f32
                            float* __restrict__ out,        // ys|cT|hT  f32
                            _Float16* __restrict__ hbuf,    // [2][B][D] f16
                            unsigned* __restrict__ counter) {
    // [gate(4)][n_local(32)][k(1024)] f16 = 256 KB of the 320 KB WGP LDS.
    extern __shared__ __align__(64) _Float16 WT[];

    const int wg   = blockIdx.x;   // owns hidden cols [wg*32, wg*32+32)
    const int tid  = threadIdx.x;
    const int lane = tid & 31;
    const int wave = tid >> 5;

    // ---- one-time: stage this WG's slice of [Wi;Wh], transposed, as f16.
    for (int i = tid; i < 4 * 32 * 1024; i += NTHREADS) {
        int k    = i >> 7;          // 0..1023 (K row: Wi then Wh)
        int c    = i & 127;
        int gate = c >> 5;
        int nl   = c & 31;
        int col  = gate * DMODEL + wg * 32 + nl;
        float w  = (k < DMODEL) ? Wi[(size_t)k * G4D + col]
                                : Wh[(size_t)(k - DMODEL) * G4D + col];
        WT[(size_t)(gate * 32 + nl) * 1024 + k] = (_Float16)w;
    }
    __syncthreads();

    // ---- tile assignment: wave -> (m_tile 0..3, local n_tile 0..1)
    const int mt     = wave & 3;
    const int ntl    = wave >> 2;
    const int m_base = mt * 16;
    const int lhalf  = (lane >> 4) & 1;          // lane group 0-15 / 16-31
    const int cl     = lane & 15;
    const int nn     = wg * 32 + ntl * 16 + cl;  // this lane's hidden column
    const int mA     = m_base + cl;              // row this lane loads for A
    const int aoff   = lhalf * 16;               // A-fragment K sub-offset
    const int boff   = lhalf * 16;               // B-fragment K sub-offset

    // Per-gate B-fragment base pointers in LDS (k folds into ds offset imm).
    const _Float16* wbase0 = WT + (size_t)(0 * 32 + ntl * 16 + cl) * 1024 + boff;
    const _Float16* wbase1 = WT + (size_t)(1 * 32 + ntl * 16 + cl) * 1024 + boff;
    const _Float16* wbase2 = WT + (size_t)(2 * 32 + ntl * 16 + cl) * 1024 + boff;
    const _Float16* wbase3 = WT + (size_t)(3 * 32 + ntl * 16 + cl) * 1024 + boff;

    // Bias is uniform per lane per gate in the C layout (all 8 elems share N).
    const float bi = bias[0 * DMODEL + nn];
    const float bf = bias[1 * DMODEL + nn];
    const float bg = bias[2 * DMODEL + nn];
    const float bo = bias[3 * DMODEL + nn];

    // Cell state for this lane's 8 C-matrix elements, lives in VGPRs all scan.
    float cst[8];
#pragma unroll
    for (int r = 0; r < 8; ++r)
        cst[r] = c0[(size_t)(m_base + lhalf * 8 + r) * DMODEL + nn];

    const size_t HB = (size_t)BATCH * DMODEL;
    const size_t YS = (size_t)T_STEPS * BATCH * DMODEL;

    for (int t = 0; t < T_STEPS; ++t) {
        const size_t    xoff   = ((size_t)t * BATCH + mA) * DMODEL + aoff;
        const float*    xrow   = x + xoff;
        const _Float16* x16row = x16 + xoff;
        const _Float16* hrow   = hbuf + (size_t)(t & 1) * HB
                                      + (size_t)mA * DMODEL + aoff;
        _Float16*       hnext  = hbuf + (size_t)((t + 1) & 1) * HB;

        // Fire-and-forget prefetch of next step's x rows (hides HBM latency
        // on the sequential critical path; no counters involved).
        if (t + 1 < T_STEPS) {
            const char* xn = PRECVT
                ? (const char*)(x16 + ((size_t)(t + 1) * BATCH + mA) * DMODEL)
                : (const char*)(x   + ((size_t)(t + 1) * BATCH + mA) * DMODEL);
            const int nfetch = PRECVT ? 4 : 8;
#pragma unroll
            for (int j = 0; j < 8; ++j)
                if (j < nfetch) __builtin_prefetch(xn + j * 256, 0, 3);
        }

        v8f acc0, acc1, acc2, acc3;
#pragma unroll
        for (int r = 0; r < 8; ++r) {
            acc0[r] = bi; acc1[r] = bf; acc2[r] = bg; acc3[r] = bo;
        }

        // ---- software-pipelined K loop over [Wi;Wh] (k = 0..1023, step 32).
        // B fragments double-buffered (loads issued one iteration ahead of
        // the WMMA group consuming them); A rotated the same way. With
        // PRECVT the A path is pure f16 b128 loads for both K regions.
        v16h a  = PRECVT ? load_a_h(x16row) : load_a_x(xrow);
        v16h b0 = *(const v16h*)(wbase0);
        v16h b1 = *(const v16h*)(wbase1);
        v16h b2 = *(const v16h*)(wbase2);
        v16h b3 = *(const v16h*)(wbase3);

#pragma unroll
        for (int k0 = 0; k0 < 2 * DMODEL; k0 += 32) {
            const bool last = (k0 + 32 >= 2 * DMODEL);
            v16h n0 = b0, n1 = b1, n2 = b2, n3 = b3, an = a;
            if (!last) {
                n0 = *(const v16h*)(wbase0 + k0 + 32);
                n1 = *(const v16h*)(wbase1 + k0 + 32);
                n2 = *(const v16h*)(wbase2 + k0 + 32);
                n3 = *(const v16h*)(wbase3 + k0 + 32);
                if (k0 + 32 < DMODEL) {
                    an = PRECVT ? load_a_h(x16row + k0 + 32)
                                : load_a_x(xrow + k0 + 32);
                } else {
                    an = load_a_h(hrow + (k0 + 32 - DMODEL));
                }
            }
            acc0 = __builtin_amdgcn_wmma_f32_16x16x32_f16(
                false, a, false, b0, (short)0, acc0, false, false);
            acc1 = __builtin_amdgcn_wmma_f32_16x16x32_f16(
                false, a, false, b1, (short)0, acc1, false, false);
            acc2 = __builtin_amdgcn_wmma_f32_16x16x32_f16(
                false, a, false, b2, (short)0, acc2, false, false);
            acc3 = __builtin_amdgcn_wmma_f32_16x16x32_f16(
                false, a, false, b3, (short)0, acc3, false, false);
            b0 = n0; b1 = n1; b2 = n2; b3 = n3; a = an;
        }

        // ---- gates + state update (lane-local: i/f/g/o share (b,d) slots)
#pragma unroll
        for (int r = 0; r < 8; ++r) {
            float si   = sigf(acc0[r]);
            float sf   = sigf(acc1[r]);
            float tg   = tanh_fast(acc2[r]);
            float so   = sigf(acc3[r]);
            float cnew = sf * cst[r] + si * tg;
            cst[r]     = cnew;
            float hh   = so * tanh_fast(cnew);
            int m      = m_base + lhalf * 8 + r;
            out[((size_t)t * BATCH + m) * DMODEL + nn] = hh;          // ys[t]
            hnext[(size_t)m * DMODEL + nn] = (_Float16)hh;            // next A
            if (t == T_STEPS - 1) {
                out[YS + (size_t)m * DMODEL + nn]      = cnew;        // cT
                out[YS + HB + (size_t)m * DMODEL + nn] = hh;          // hT
            }
        }

        // ---- device-wide barrier between time steps (monotonic counter)
        __syncthreads();
        if (tid == 0) {
            __hip_atomic_fetch_add(counter, 1u, __ATOMIC_RELEASE,
                                   __HIP_MEMORY_SCOPE_AGENT);
            const unsigned target = (unsigned)NWG * (unsigned)(t + 1);
            while (__hip_atomic_load(counter, __ATOMIC_ACQUIRE,
                                     __HIP_MEMORY_SCOPE_AGENT) < target) {
                __builtin_amdgcn_s_sleep(1);
            }
        }
        __syncthreads();
        __threadfence();
    }
}

extern "C" void kernel_launch(void* const* d_in, const int* in_sizes, int n_in,
                              void* d_out, int out_size, void* d_ws, size_t ws_size,
                              hipStream_t stream) {
    (void)in_sizes; (void)n_in; (void)out_size;
    const float* x  = (const float*)d_in[0];  // inputs [T,B,D]
    const float* c0 = (const float*)d_in[1];  // [B,D]
    const float* h0 = (const float*)d_in[2];  // [B,D]
    const float* Wi = (const float*)d_in[3];  // [D,4D]
    const float* Wh = (const float*)d_in[4];  // [D,4D]
    const float* b  = (const float*)d_in[5];  // [4D]
    float* out = (float*)d_out;

    // Workspace layout:
    //   [0,256)            barrier counter (padded)
    //   [256, 256+128K)    h f16 ping-pong  (2 x 64 x 512 f16)
    //   [131328, +64MB)    optional x f16 copy
    unsigned* counter = (unsigned*)d_ws;
    _Float16* hbuf    = (_Float16*)((char*)d_ws + 256);
    _Float16* x16     = (_Float16*)((char*)d_ws + 131328);

    const size_t xelems   = (size_t)T_STEPS * BATCH * DMODEL;
    const size_t ws_need  = 131328 + xelems * sizeof(_Float16);
    const bool   precvt   = (ws_size >= ws_need);

    lstm_init_kernel<<<(BATCH * DMODEL + 255) / 256, 256, 0, stream>>>(
        h0, hbuf, counter);

    const size_t lds_bytes = (size_t)4 * 32 * 1024 * sizeof(_Float16); // 256 KB

    if (precvt) {
        lstm_xcvt_kernel<<<4096, 256, 0, stream>>>(x, x16, (long)(xelems / 4));
        lstm_persistent_kernel<true><<<NWG, NTHREADS, lds_bytes, stream>>>(
            x, x16, c0, Wi, Wh, b, out, hbuf, counter);
    } else {
        lstm_persistent_kernel<false><<<NWG, NTHREADS, lds_bytes, stream>>>(
            x, x16, c0, Wi, Wh, b, out, hbuf, counter);
    }
}
